// RPE_GNN_36455682409152
// MI455X (gfx1250) — matrix operand order repaired and checked by reading, not compiled, not run on hardware
//
#include <hip/hip_runtime.h>
#include <hip/hip_bf16.h>
#include <math.h>

typedef __bf16 bf16;
typedef bf16  v16bf __attribute__((ext_vector_type(16)));
typedef bf16  v8bf  __attribute__((ext_vector_type(8)));
typedef float v8f   __attribute__((ext_vector_type(8)));

#define N_NODES 100000
#define N_FEAT  128
#define HID     256
#define NCLS    40
#define BATCH   4096
#define PMAX    331776
#define LN_EPS  1e-5f

// ---------------- WMMA fragment helpers (contiguous k = 16*(lane/16) + s in both A and B) ----

// A fragment 16x32 (MxK) from row-major bf16 tile in LDS. base points at (row0, k0).
__device__ __forceinline__ v16bf frag_a_lds(const bf16* base, int ld, int lane) {
    int hi = (lane >> 4) & 1;      // lanes 0-15: K 0..15 ; lanes 16-31: K 16..31
    int m  = lane & 15;
    const bf16* p = base + m * ld + hi * 16;
    v16bf r;
    ((v8bf*)&r)[0] = *(const v8bf*)(p);        // K = hi*16 .. hi*16+7
    ((v8bf*)&r)[1] = *(const v8bf*)(p + 8);    // K = hi*16+8 .. +7
    return r;
}

// B fragment 32x16 (KxN) from weights stored TRANSPOSED as [N][K] bf16 (contiguous K).
__device__ __forceinline__ v16bf frag_b_nk(const bf16* wT, int ldk, int n0, int k0, int lane) {
    int hi = (lane >> 4) & 1;      // lanes 0-15: K 0..15 ; lanes 16-31: K 16..31
    int n  = lane & 15;
    return *(const v16bf*)(wT + (size_t)(n0 + n) * ldk + (k0 + hi * 16));
}

// ---------------- small utility kernels --------------------------------------------

// src f32 [L][K][N] -> dst bf16 [L][N][K]
__global__ void k_transpose_bf16(const float* __restrict__ src, bf16* __restrict__ dst,
                                 int L, int K, int N) {
    long idx = (long)blockIdx.x * blockDim.x + threadIdx.x;
    long total = (long)L * K * N;
    if (idx >= total) return;
    int l = (int)(idx / ((long)K * N));
    int r = (int)(idx % ((long)K * N));
    int kk = r / N, n = r % N;
    dst[(long)l * K * N + (long)n * K + kk] = (bf16)src[idx];
}

__global__ void k_init_zero(bf16* __restrict__ emb0, float* __restrict__ acc) {
    int idx = blockIdx.x * blockDim.x + threadIdx.x;
    if (idx < BATCH * HID)  emb0[idx] = (bf16)0.0f;
    if (idx < BATCH * NCLS) acc[idx]  = 0.0f;
}

// pred_features = nf @ pw + pb   (f32, tiny: 1 GFLOP, VALU is fine)
__global__ void k_pred_gemm(const float* __restrict__ nf, const float* __restrict__ pw,
                            const float* __restrict__ pb, float* __restrict__ out) {
    int idx = blockIdx.x * blockDim.x + threadIdx.x;
    if (idx >= N_NODES * NCLS) return;
    int n = idx / NCLS, c = idx % NCLS;
    float s = pb[c];
    const float* row = nf + (size_t)n * N_FEAT;
#pragma unroll 4
    for (int k = 0; k < N_FEAT; ++k) s = fmaf(row[k], pw[k * NCLS + c], s);
    out[idx] = s;
}

// ---------------- features = nf @ tw + tb -> bf16   (WMMA, BM=64 BN=256 K=128) ------

__global__ __launch_bounds__(256) void k_feat_gemm(const float* __restrict__ nf,
                                                   const bf16* __restrict__ twT,
                                                   const float* __restrict__ tb,
                                                   bf16* __restrict__ feat) {
    __shared__ __align__(16) bf16 As[64 * N_FEAT];   // 16 KB
    int tid = threadIdx.x;
    int row0 = blockIdx.x * 64;

    { // stage A: 4 threads/row, 32 cols each, f32 -> bf16
        int r = tid >> 2, seg = tid & 3, c0 = seg * 32;
        int rg = row0 + r;
        if (rg < N_NODES) {
            const float4* p = (const float4*)(nf + (size_t)rg * N_FEAT + c0);
            bf16* d = As + r * N_FEAT + c0;
#pragma unroll
            for (int i = 0; i < 8; ++i) {
                float4 v = p[i];
                d[i*4+0] = (bf16)v.x; d[i*4+1] = (bf16)v.y;
                d[i*4+2] = (bf16)v.z; d[i*4+3] = (bf16)v.w;
            }
        } else {
            bf16* d = As + r * N_FEAT + c0;
            for (int i = 0; i < 32; ++i) d[i] = (bf16)0.0f;
        }
    }
    __syncthreads();

    int wave = tid >> 5, lane = tid & 31;
    int wm = wave >> 2, wn = wave & 3;          // 2 x 4 wave grid, 32x64 per wave
    v8f acc[2][4];
#pragma unroll
    for (int i = 0; i < 2; ++i)
#pragma unroll
        for (int j = 0; j < 4; ++j) acc[i][j] = (v8f){0,0,0,0,0,0,0,0};

    for (int k0 = 0; k0 < N_FEAT; k0 += 32) {
        v16bf a[2];
#pragma unroll
        for (int i = 0; i < 2; ++i)
            a[i] = frag_a_lds(As + (wm * 32 + i * 16) * N_FEAT + k0, N_FEAT, lane);
#pragma unroll
        for (int j = 0; j < 4; ++j) {
            v16bf b = frag_b_nk(twT, N_FEAT, wn * 64 + j * 16, k0, lane);
#pragma unroll
            for (int i = 0; i < 2; ++i)
                acc[i][j] = __builtin_amdgcn_wmma_f32_16x16x32_bf16(
                    false, a[i], false, b, (short)0, acc[i][j], false, false);
        }
    }

    int hi = lane >> 4, nl = lane & 15;
#pragma unroll
    for (int i = 0; i < 2; ++i)
#pragma unroll
        for (int j = 0; j < 4; ++j)
#pragma unroll
            for (int v = 0; v < 8; ++v) {
                int rg = row0 + wm * 32 + i * 16 + v + 8 * hi;
                int c  = wn * 64 + j * 16 + nl;
                if (rg < N_NODES)
                    feat[(size_t)rg * HID + c] = (bf16)(acc[i][j][v] + tb[c]);
            }
}

// ---------------- fused layer: gather-diff + dual WMMA GEMM + ReLU + LayerNorm ------
// out[i] = LN( relu((F[child[i]]-F[parent[i]]) @ w1 + b1) + relu(E[i/k] @ w2 + b2) )

__global__ __launch_bounds__(256) void k_layer(
    const bf16* __restrict__ feat, const int* __restrict__ child,
    const int* __restrict__ prev_ids, const bf16* __restrict__ prev_embed,
    int kexp, int P,
    const bf16* __restrict__ w1T, const float* __restrict__ b1,
    const bf16* __restrict__ w2T, const float* __restrict__ b2,
    const float* __restrict__ lng, const float* __restrict__ lnb,
    bf16* __restrict__ out_embed)
{
    __shared__ __align__(16) unsigned char smem[64 * HID * 4];   // 64 KB
    bf16*  A1  = (bf16*)smem;                 // 64x256 bf16 (diff)   32 KB
    bf16*  A2  = (bf16*)(smem + 32768);       // 64x256 bf16 (embed)  32 KB
    float* OUT = (float*)smem;                // aliased after GEMM: 64x256 f32

    int tid = threadIdx.x;
    int row0 = blockIdx.x * 64;

    { // phase 0: stage A1 = F[child]-F[parent], A2 = prev_embed[row/k]
        int r = tid >> 2, seg = tid & 3, c0 = seg * 64;
        int rg = row0 + r;
        uint4* d1 = (uint4*)(A1 + r * HID + c0);
        uint4* d2 = (uint4*)(A2 + r * HID + c0);
        if (rg < P) {
            int ch = child[rg];
            int pr = rg / kexp;
            int pa = prev_ids[pr];
            const uint4* pc = (const uint4*)(feat + (size_t)ch * HID + c0);
            const uint4* pp = (const uint4*)(feat + (size_t)pa * HID + c0);
            const uint4* pe = (const uint4*)(prev_embed + (size_t)pr * HID + c0);
#pragma unroll
            for (int i = 0; i < 8; ++i) {
                uint4 vc = pc[i], vp = pp[i], o;
                const bf16* bc = (const bf16*)&vc;
                const bf16* bp = (const bf16*)&vp;
                bf16* bo = (bf16*)&o;
#pragma unroll
                for (int j = 0; j < 8; ++j)
                    bo[j] = (bf16)((float)bc[j] - (float)bp[j]);
                d1[i] = o;
                d2[i] = pe[i];
            }
        } else {
            uint4 z = {0, 0, 0, 0};
            for (int i = 0; i < 8; ++i) { d1[i] = z; d2[i] = z; }
        }
    }
    __syncthreads();

    // phase 1: dual GEMM (two accumulator sets, independent ReLU paths)
    int wave = tid >> 5, lane = tid & 31;
    int wm = wave >> 2, wn = wave & 3;          // 2 x 4 waves, 32x64 tile per wave
    v8f acc1[2][4], acc2[2][4];
#pragma unroll
    for (int i = 0; i < 2; ++i)
#pragma unroll
        for (int j = 0; j < 4; ++j) {
            acc1[i][j] = (v8f){0,0,0,0,0,0,0,0};
            acc2[i][j] = (v8f){0,0,0,0,0,0,0,0};
        }

    for (int k0 = 0; k0 < HID; k0 += 32) {
        v16bf a1[2], a2[2];
#pragma unroll
        for (int i = 0; i < 2; ++i) {
            a1[i] = frag_a_lds(A1 + (wm * 32 + i * 16) * HID + k0, HID, lane);
            a2[i] = frag_a_lds(A2 + (wm * 32 + i * 16) * HID + k0, HID, lane);
        }
#pragma unroll
        for (int j = 0; j < 4; ++j) {
            v16bf bb1 = frag_b_nk(w1T, HID, wn * 64 + j * 16, k0, lane);
            v16bf bb2 = frag_b_nk(w2T, HID, wn * 64 + j * 16, k0, lane);
#pragma unroll
            for (int i = 0; i < 2; ++i) {
                acc1[i][j] = __builtin_amdgcn_wmma_f32_16x16x32_bf16(
                    false, a1[i], false, bb1, (short)0, acc1[i][j], false, false);
                acc2[i][j] = __builtin_amdgcn_wmma_f32_16x16x32_bf16(
                    false, a2[i], false, bb2, (short)0, acc2[i][j], false, false);
            }
        }
    }
    __syncthreads();    // done reading A1/A2; OUT aliases the same LDS

    { // phase 2: bias + relu + sum -> f32 LDS
        int hi = lane >> 4, nl = lane & 15;
#pragma unroll
        for (int i = 0; i < 2; ++i)
#pragma unroll
            for (int j = 0; j < 4; ++j)
#pragma unroll
                for (int v = 0; v < 8; ++v) {
                    int rr = wm * 32 + i * 16 + v + 8 * hi;
                    int c  = wn * 64 + j * 16 + nl;
                    float x1 = fmaxf(acc1[i][j][v] + b1[c], 0.0f);
                    float x2 = fmaxf(acc2[i][j][v] + b2[c], 0.0f);
                    OUT[rr * HID + c] = x1 + x2;
                }
    }
    __syncthreads();

    { // phase 3: LayerNorm across 256 cols (4 threads/row, shfl reduce) + bf16 store
        int r = tid >> 2, seg = tid & 3, c0 = seg * 64;
        int rg = row0 + r;
        float s = 0.f, ss = 0.f;
        const float* rowp = OUT + r * HID;
#pragma unroll 8
        for (int c = c0; c < c0 + 64; ++c) { float x = rowp[c]; s += x; ss += x * x; }
        s  += __shfl_xor(s, 1);  s  += __shfl_xor(s, 2);
        ss += __shfl_xor(ss, 1); ss += __shfl_xor(ss, 2);
        float mu  = s * (1.0f / HID);
        float var = ss * (1.0f / HID) - mu * mu;
        float inv = rsqrtf(var + LN_EPS);
        if (rg < P) {
            bf16* dst = out_embed + (size_t)rg * HID;
            for (int cc = 0; cc < 64; cc += 8) {
                uint4 pack;
                bf16* bo = (bf16*)&pack;
#pragma unroll
                for (int j = 0; j < 8; ++j) {
                    int c = c0 + cc + j;
                    bo[j] = (bf16)((rowp[c] - mu) * inv * lng[c] + lnb[c]);
                }
                *(uint4*)(dst + c0 + cc) = pack;
            }
        }
    }
}

// ---------------- layer 7 (N=40, no LN) fused with multiply / L2-norm / segment-sum -

__global__ __launch_bounds__(256) void k_last_layer(
    const bf16* __restrict__ feat, const int* __restrict__ child,
    const int* __restrict__ prev_ids, const bf16* __restrict__ prev_embed,
    int P,
    const bf16* __restrict__ w1T, const float* __restrict__ b1,
    const bf16* __restrict__ w2T, const float* __restrict__ b2,
    const float* __restrict__ predf, float* __restrict__ emb_acc)
{
    const int RPB = 32;
    __shared__ __align__(16) bf16 D[RPB * HID];    // 16 KB (diff)
    __shared__ __align__(16) bf16 E[RPB * HID];    // 16 KB (embed)
    __shared__ float F[RPB * NCLS];
    __shared__ float INV[RPB];

    int tid = threadIdx.x;
    int row0 = blockIdx.x * RPB;

    { // stage: 8 threads/row, 32 cols each
        int r = tid >> 3, seg = tid & 7, c0 = seg * 32;
        int rg = row0 + r;
        uint4* d1 = (uint4*)(D + r * HID + c0);
        uint4* d2 = (uint4*)(E + r * HID + c0);
        if (rg < P) {
            int ch = child[rg];
            int pa = prev_ids[rg];                 // k == 1
            const uint4* pc = (const uint4*)(feat + (size_t)ch * HID + c0);
            const uint4* pp = (const uint4*)(feat + (size_t)pa * HID + c0);
            const uint4* pe = (const uint4*)(prev_embed + (size_t)rg * HID + c0);
#pragma unroll
            for (int i = 0; i < 4; ++i) {
                uint4 vc = pc[i], vp = pp[i], o;
                const bf16* bc = (const bf16*)&vc;
                const bf16* bp = (const bf16*)&vp;
                bf16* bo = (bf16*)&o;
#pragma unroll
                for (int j = 0; j < 8; ++j)
                    bo[j] = (bf16)((float)bc[j] - (float)bp[j]);
                d1[i] = o;
                d2[i] = pe[i];
            }
        } else {
            uint4 z = {0, 0, 0, 0};
            for (int i = 0; i < 4; ++i) { d1[i] = z; d2[i] = z; }
        }
    }
    __syncthreads();

    // 32x40 outputs, 5 per thread: o = relu(d@w1+b1)+relu(e@w2+b2); f = o * predf[child]
    for (int t = tid; t < RPB * NCLS; t += 256) {
        int r = t / NCLS, c = t % NCLS;
        float s1 = b1[c], s2 = b2[c];
        const bf16* w1c = w1T + (size_t)c * HID;
        const bf16* w2c = w2T + (size_t)c * HID;
        const bf16* dr = D + r * HID;
        const bf16* er = E + r * HID;
        for (int kk = 0; kk < HID; kk += 8) {
            v8bf dv  = *(const v8bf*)(dr + kk);
            v8bf ev  = *(const v8bf*)(er + kk);
            v8bf wv1 = *(const v8bf*)(w1c + kk);
            v8bf wv2 = *(const v8bf*)(w2c + kk);
#pragma unroll
            for (int j = 0; j < 8; ++j) {
                s1 = fmaf((float)dv[j], (float)wv1[j], s1);
                s2 = fmaf((float)ev[j], (float)wv2[j], s2);
            }
        }
        float o = fmaxf(s1, 0.0f) + fmaxf(s2, 0.0f);
        int rg = row0 + r;
        float pf = (rg < P) ? predf[(size_t)child[rg] * NCLS + c] : 0.0f;
        F[t] = o * pf;
    }
    __syncthreads();

    if (tid < RPB) {
        float ss = 0.f;
        for (int c = 0; c < NCLS; ++c) { float x = F[tid * NCLS + c]; ss += x * x; }
        float nrm = sqrtf(ss);
        INV[tid] = (nrm == 0.0f) ? 1.0f : (1.0f / nrm);
    }
    __syncthreads();

    for (int t = tid; t < RPB * NCLS; t += 256) {
        int r = t / NCLS, c = t % NCLS;
        int rg = row0 + r;
        if (rg < P)
            atomicAdd(&emb_acc[(size_t)(rg / 81) * NCLS + c], F[t] * INV[r]);
    }
}

// ---------------- finalize: /81 + pred gather + log_softmax -------------------------

__global__ void k_finalize(const float* __restrict__ emb_acc, const float* __restrict__ predf,
                           const int* __restrict__ nodes_batch, float* __restrict__ out) {
    int r = blockIdx.x * blockDim.x + threadIdx.x;
    if (r >= BATCH) return;
    int nb = nodes_batch[r];
    float e[NCLS];
    float mx = -INFINITY;
#pragma unroll
    for (int c = 0; c < NCLS; ++c) {
        e[c] = emb_acc[r * NCLS + c] * (1.0f / 81.0f) + predf[(size_t)nb * NCLS + c];
        mx = fmaxf(mx, e[c]);
    }
    float se = 0.f;
#pragma unroll
    for (int c = 0; c < NCLS; ++c) se += expf(e[c] - mx);
    float lse = mx + logf(se);
#pragma unroll
    for (int c = 0; c < NCLS; ++c) {
        out[r * NCLS + c] = e[c];
        out[BATCH * NCLS + r * NCLS + c] = e[c] - lse;
    }
}

// ---------------- host driver -------------------------------------------------------

extern "C" void kernel_launch(void* const* d_in, const int* in_sizes, int n_in,
                              void* d_out, int out_size, void* d_ws, size_t ws_size,
                              hipStream_t stream) {
    const float* node_features = (const float*)d_in[0];
    const float* tw       = (const float*)d_in[1];
    const float* tb       = (const float*)d_in[2];
    const float* pw       = (const float*)d_in[3];
    const float* pb       = (const float*)d_in[4];
    const float* w1_stack = (const float*)d_in[5];
    const float* b1_stack = (const float*)d_in[6];
    const float* w2_stack = (const float*)d_in[7];
    const float* b2_stack = (const float*)d_in[8];
    const float* w1_last  = (const float*)d_in[9];
    const float* b1_last  = (const float*)d_in[10];
    const float* w2_last  = (const float*)d_in[11];
    const float* b2_last  = (const float*)d_in[12];
    const float* ln_g     = (const float*)d_in[13];
    const float* ln_b     = (const float*)d_in[14];
    const int* nodes_batch = (const int*)d_in[15];
    const int* neighs[8];
    for (int i = 0; i < 8; ++i) neighs[i] = (const int*)d_in[16 + i];
    (void)in_sizes; (void)n_in; (void)out_size; (void)ws_size;

    char* ws = (char*)d_ws;
    size_t off = 0;
    auto carve = [&](size_t bytes) -> void* {
        void* p = ws + off;
        off += (bytes + 255) & ~(size_t)255;
        return p;
    };
    bf16*  featB  = (bf16*) carve((size_t)N_NODES * HID * 2);   // 51.2 MB (L2-resident)
    float* predF  = (float*)carve((size_t)N_NODES * NCLS * 4);  // 16 MB
    bf16*  twT    = (bf16*) carve((size_t)HID * N_FEAT * 2);
    bf16*  w1T    = (bf16*) carve((size_t)7 * HID * HID * 2);
    bf16*  w2T    = (bf16*) carve((size_t)7 * HID * HID * 2);
    bf16*  w1Tl   = (bf16*) carve((size_t)NCLS * HID * 2);
    bf16*  w2Tl   = (bf16*) carve((size_t)NCLS * HID * 2);
    float* embAcc = (float*)carve((size_t)BATCH * NCLS * 4);
    bf16*  embA   = (bf16*) carve((size_t)PMAX * HID * 2);      // 170 MB
    bf16*  embB   = (bf16*) carve((size_t)PMAX * HID * 2);      // 170 MB

    // weight convert/transpose -> bf16 [N][K]
    auto grid1 = [](long t) { return dim3((unsigned)((t + 255) / 256)); };
    k_transpose_bf16<<<grid1((long)N_FEAT * HID), 256, 0, stream>>>(tw, twT, 1, N_FEAT, HID);
    k_transpose_bf16<<<grid1((long)7 * HID * HID), 256, 0, stream>>>(w1_stack, w1T, 7, HID, HID);
    k_transpose_bf16<<<grid1((long)7 * HID * HID), 256, 0, stream>>>(w2_stack, w2T, 7, HID, HID);
    k_transpose_bf16<<<grid1((long)HID * NCLS), 256, 0, stream>>>(w1_last, w1Tl, 1, HID, NCLS);
    k_transpose_bf16<<<grid1((long)HID * NCLS), 256, 0, stream>>>(w2_last, w2Tl, 1, HID, NCLS);

    k_init_zero<<<grid1((long)BATCH * HID), 256, 0, stream>>>(embA, embAcc);

    k_feat_gemm<<<dim3((N_NODES + 63) / 64), 256, 0, stream>>>(node_features, twT, tb, featB);
    k_pred_gemm<<<grid1((long)N_NODES * NCLS), 256, 0, stream>>>(node_features, pw, pb, predF);

    const int P[8] = {12288, 36864, 110592, 331776, 331776, 331776, 331776, 331776};
    const int KE[8] = {3, 3, 3, 3, 1, 1, 1, 1};
    bf16* ein = embA;
    bf16* eout = embB;
    for (int l = 0; l < 7; ++l) {
        const int* prev = (l == 0) ? nodes_batch : neighs[l - 1];
        k_layer<<<dim3(P[l] / 64), 256, 0, stream>>>(
            featB, neighs[l], prev, ein, KE[l], P[l],
            w1T + (size_t)l * HID * HID, b1_stack + l * HID,
            w2T + (size_t)l * HID * HID, b2_stack + l * HID,
            ln_g + l * HID, ln_b + l * HID, eout);
        bf16* t = ein; ein = eout; eout = t;
    }

    k_last_layer<<<dim3(P[7] / 32), 256, 0, stream>>>(
        featB, neighs[7], neighs[6], ein, P[7],
        w1Tl, b1_last, w2Tl, b2_last, predF, embAcc);

    k_finalize<<<dim3((BATCH + 255) / 256), 256, 0, stream>>>(embAcc, predF, nodes_batch,
                                                              (float*)d_out);
}
